// MultiHeadAttention_25297357373877
// MI455X (gfx1250) — compile-verified, matrix operand order checked by reading, not run on hardware
//
#include <hip/hip_runtime.h>
#include <hip/hip_bf16.h>

#define S_LEN   2048
#define NHEAD   16
#define DKH     64
#define DMODEL  1024
#define BATCH   2

typedef __attribute__((ext_vector_type(16))) __bf16         v16bf;
typedef __attribute__((ext_vector_type(8)))  float          v8f;
typedef __attribute__((ext_vector_type(8)))  unsigned short us8;
typedef __attribute__((ext_vector_type(16))) unsigned short us16;

// ---------- helpers ----------

__device__ __forceinline__ unsigned short f2bf(float f) {
  unsigned int u = __builtin_bit_cast(unsigned int, f);
  u += 0x7FFFu + ((u >> 16) & 1u);            // round-to-nearest-even
  return (unsigned short)(u >> 16);
}

// Load one 16-bit WMMA A/B fragment (16 bf16 values per lane).
// Per ISA layout: lanes 0-15 hold k-offsets {0..7, 16..23}, lanes 16-31 hold
// {8..15, 24..31} (relative to kbase). Two aligned 16B loads.
__device__ __forceinline__ v16bf load_frag(const unsigned short* rowptr, int kbase) {
  int lane = threadIdx.x & 31;
  int lh   = (lane >> 4) << 3;                // 0 or 8
  const us8* p0 = reinterpret_cast<const us8*>(rowptr + kbase + lh);
  const us8* p1 = reinterpret_cast<const us8*>(rowptr + kbase + 16 + lh);
  us8 lo = *p0;
  us8 hi = *p1;
  us16 u = __builtin_shufflevector(lo, hi, 0,1,2,3,4,5,6,7,8,9,10,11,12,13,14,15);
  return __builtin_bit_cast(v16bf, u);
}

__device__ __forceinline__ v8f wmma_bf16(v16bf a, v16bf b, v8f c) {
  return __builtin_amdgcn_wmma_f32_16x16x32_bf16(
      /*neg_a=*/false, a, /*neg_b=*/false, b,
      /*c_mod=*/(short)0, c, /*reuse_a=*/false, /*reuse_b=*/false);
}

// ---------- kernels ----------

__global__ void k_f32_to_bf16(const float* __restrict__ in,
                              unsigned short* __restrict__ out, int n) {
  int i = blockIdx.x * blockDim.x + threadIdx.x;
  if (i < n) out[i] = f2bf(in[i]);
}

// W[k][n] (f32, 1024x1024) -> Wt[n][k] (bf16), LDS tiled transpose
__global__ void k_transpose_bf16(const float* __restrict__ W,
                                 unsigned short* __restrict__ Wt) {
  __shared__ unsigned short tile[32][33];
  int tx = threadIdx.x, ty = threadIdx.y;     // 32 x 8
  int c = blockIdx.x * 32 + tx;
#pragma unroll
  for (int i = 0; i < 4; i++) {
    int r = blockIdx.y * 32 + ty + i * 8;
    tile[ty + i * 8][tx] = f2bf(W[(size_t)r * DMODEL + c]);
  }
  __syncthreads();
  int r2 = blockIdx.y * 32 + tx;
#pragma unroll
  for (int i = 0; i < 4; i++) {
    int c2 = blockIdx.x * 32 + ty + i * 8;
    Wt[(size_t)c2 * DMODEL + r2] = tile[tx][ty + i * 8];
  }
}

// C[M,N] = A[M,K] * Bt[N,K]^T + bias.  A,Bt bf16 (K = N = DMODEL).
// Per-wave tile 16(M) x 64(N), block = 4 waves stacked along M.
// mode 0: write bf16 head-split [B,H,S,DKH]
// mode 1: write bf16 head-split transposed [B,H,DKH,S]
// mode 2: write f32 row-major [M, DMODEL]
__global__ void k_gemm_wmma(const unsigned short* __restrict__ A,
                            const unsigned short* __restrict__ Bt,
                            const float* __restrict__ bias,
                            void* __restrict__ outp, int mode) {
  int lane = threadIdx.x & 31;
  int laneLo = lane & 15;
  int lh = (lane >> 4) << 3;
  int wave = threadIdx.x >> 5;
  int n0 = blockIdx.x * 64;
  int m0 = blockIdx.y * 64 + wave * 16;

  const unsigned short* arow = A + (size_t)(m0 + laneLo) * DMODEL;
  const unsigned short* bcol[4];
#pragma unroll
  for (int t = 0; t < 4; t++)
    bcol[t] = Bt + (size_t)(n0 + t * 16 + laneLo) * DMODEL;

  v8f acc[4] = {};
  for (int kb = 0; kb < DMODEL; kb += 32) {
    v16bf af = load_frag(arow, kb);
#pragma unroll
    for (int t = 0; t < 4; t++)
      acc[t] = wmma_bf16(af, load_frag(bcol[t], kb), acc[t]);
  }

#pragma unroll
  for (int t = 0; t < 4; t++) {
    int n = n0 + t * 16 + laneLo;
    float bv = bias[n];
#pragma unroll
    for (int j = 0; j < 8; j++) {
      int m = m0 + j + lh;                    // global row in [0, B*S)
      float v = acc[t][j] + bv;
      if (mode == 2) {
        ((float*)outp)[(size_t)m * DMODEL + n] = v;
      } else {
        int b = m >> 11, s = m & (S_LEN - 1);
        int h = n >> 6, d = n & (DKH - 1);
        unsigned short* o = (unsigned short*)outp;
        if (mode == 0)
          o[((size_t)(b * NHEAD + h) * S_LEN + s) * DKH + d] = f2bf(v);
        else
          o[((size_t)(b * NHEAD + h) * DKH + d) * S_LEN + s] = f2bf(v);
      }
    }
  }
}

// Flash attention: Q,K [B,H,S,64] bf16; Vt [B,H,64,S] bf16; mask [B,S] i32.
// One wave handles 16 query rows; block = 4 waves; 32 keys per iteration.
// Output written bf16 merged [B*S, DMODEL] for the final projection.
__global__ void k_flash_attn(const unsigned short* __restrict__ Q,
                             const unsigned short* __restrict__ K,
                             const unsigned short* __restrict__ Vt,
                             const int* __restrict__ mask,
                             unsigned short* __restrict__ O) {
  __shared__ __align__(16) unsigned short ldsP[4][16 * 32];
  int lane = threadIdx.x & 31, laneLo = lane & 15;
  int lh = (lane >> 4) << 3;
  int wave = threadIdx.x >> 5;
  int b = blockIdx.z, h = blockIdx.y;
  int q0 = blockIdx.x * 64 + wave * 16;

  size_t headBase = (size_t)(b * NHEAD + h) * S_LEN * DKH;
  const unsigned short* qrow = Q + headBase + (size_t)(q0 + laneLo) * DKH;
  v16bf qa0 = load_frag(qrow, 0);
  v16bf qa1 = load_frag(qrow, 32);

  v8f o0 = {}, o1 = {}, o2 = {}, o3 = {};
  float mi[8], li[8];
#pragma unroll
  for (int j = 0; j < 8; j++) { mi[j] = -3.0e38f; li[j] = 0.f; }

  const unsigned short* vbase = Vt + headBase;   // [64][S]
  unsigned short* myP = ldsP[wave];

  for (int kb = 0; kb < S_LEN; kb += 32) {
    // two 16-key score tiles (d_k = 64 -> 2 WMMAs each)
    v8f s0 = {}, s1 = {};
    {
      const unsigned short* krow = K + headBase + (size_t)(kb + laneLo) * DKH;
      s0 = wmma_bf16(qa0, load_frag(krow, 0), s0);
      s0 = wmma_bf16(qa1, load_frag(krow, 32), s0);
    }
    {
      const unsigned short* krow = K + headBase + (size_t)(kb + 16 + laneLo) * DKH;
      s1 = wmma_bf16(qa0, load_frag(krow, 0), s1);
      s1 = wmma_bf16(qa1, load_frag(krow, 32), s1);
    }
    int mk0 = mask[b * S_LEN + kb + laneLo];
    int mk1 = mask[b * S_LEN + kb + 16 + laneLo];

#pragma unroll
    for (int j = 0; j < 8; j++) {
      float a = s0[j] * 0.125f; if (mk0 == 0) a = -1.0e9f;
      float c = s1[j] * 0.125f; if (mk1 == 0) c = -1.0e9f;
      float mx = fmaxf(a, c);
#pragma unroll
      for (int d = 1; d < 16; d <<= 1) mx = fmaxf(mx, __shfl_xor(mx, d, 32));
      float mnew = fmaxf(mi[j], mx);
      float corr = __expf(mi[j] - mnew);
      mi[j] = mnew;
      float p0 = __expf(a - mnew);
      float p1 = __expf(c - mnew);
      float rs = p0 + p1;
#pragma unroll
      for (int d = 1; d < 16; d <<= 1) rs += __shfl_xor(rs, d, 32);
      li[j] = li[j] * corr + rs;
      o0[j] *= corr; o1[j] *= corr; o2[j] *= corr; o3[j] *= corr;
      // stage P tile (16x32) to LDS as bf16, C-fragment layout -> row major
      myP[(j + lh) * 32 + laneLo]      = f2bf(p0);
      myP[(j + lh) * 32 + 16 + laneLo] = f2bf(p1);
    }
    __syncthreads();

    v16bf pa = load_frag(myP + laneLo * 32, 0);
    o0 = wmma_bf16(pa, load_frag(vbase + (size_t)(0 * 16 + laneLo) * S_LEN, kb), o0);
    o1 = wmma_bf16(pa, load_frag(vbase + (size_t)(1 * 16 + laneLo) * S_LEN, kb), o1);
    o2 = wmma_bf16(pa, load_frag(vbase + (size_t)(2 * 16 + laneLo) * S_LEN, kb), o2);
    o3 = wmma_bf16(pa, load_frag(vbase + (size_t)(3 * 16 + laneLo) * S_LEN, kb), o3);
    __syncthreads();
  }

#pragma unroll
  for (int j = 0; j < 8; j++) {
    float inv = 1.0f / li[j];
    int row = q0 + j + lh;
    size_t base = ((size_t)b * S_LEN + row) * DMODEL + h * DKH + laneLo;
    O[base + 0]  = f2bf(o0[j] * inv);
    O[base + 16] = f2bf(o1[j] * inv);
    O[base + 32] = f2bf(o2[j] * inv);
    O[base + 48] = f2bf(o3[j] * inv);
  }
}

// ---------- launch ----------

extern "C" void kernel_launch(void* const* d_in, const int* in_sizes, int n_in,
                              void* d_out, int out_size, void* d_ws, size_t ws_size,
                              hipStream_t stream) {
  (void)in_sizes; (void)n_in; (void)out_size; (void)ws_size;
  const float* q    = (const float*)d_in[0];
  const float* k    = (const float*)d_in[1];
  const float* v    = (const float*)d_in[2];
  const int*   mask = (const int*)  d_in[3];
  const float* Wq = (const float*)d_in[4];  const float* bq = (const float*)d_in[5];
  const float* Wk = (const float*)d_in[6];  const float* bk = (const float*)d_in[7];
  const float* Wv = (const float*)d_in[8];  const float* bv = (const float*)d_in[9];
  const float* Wo = (const float*)d_in[10]; const float* bo = (const float*)d_in[11];

  const size_t MTOT = (size_t)BATCH * S_LEN;         // 4096
  const size_t ACT  = MTOT * DMODEL;                 // 4 Mi elements
  const size_t WSZ  = (size_t)DMODEL * DMODEL;       // 1 Mi elements

  unsigned short* ws  = (unsigned short*)d_ws;
  unsigned short* Xq  = ws;
  unsigned short* Xk  = Xq + ACT;
  unsigned short* Xv  = Xk + ACT;
  unsigned short* Wtq = Xv + ACT;
  unsigned short* Wtk = Wtq + WSZ;
  unsigned short* Wtv = Wtk + WSZ;
  unsigned short* Wto = Wtv + WSZ;
  unsigned short* Qh  = Wto + WSZ;
  unsigned short* Kh  = Qh + ACT;
  unsigned short* Vh  = Kh + ACT;
  unsigned short* AO  = Vh + ACT;                    // total 64 MiB

  // 1) activations -> bf16
  int cvtBlocks = (int)(ACT / 256);
  k_f32_to_bf16<<<cvtBlocks, 256, 0, stream>>>(q, Xq, (int)ACT);
  k_f32_to_bf16<<<cvtBlocks, 256, 0, stream>>>(k, Xk, (int)ACT);
  k_f32_to_bf16<<<cvtBlocks, 256, 0, stream>>>(v, Xv, (int)ACT);

  // 2) weights -> bf16 transposed
  dim3 tb(32, 8), tg(DMODEL / 32, DMODEL / 32);
  k_transpose_bf16<<<tg, tb, 0, stream>>>(Wq, Wtq);
  k_transpose_bf16<<<tg, tb, 0, stream>>>(Wk, Wtk);
  k_transpose_bf16<<<tg, tb, 0, stream>>>(Wv, Wtv);
  k_transpose_bf16<<<tg, tb, 0, stream>>>(Wo, Wto);

  // 3) Q/K/V projections (WMMA)
  dim3 gb(128), gg(DMODEL / 64, (unsigned)(MTOT / 64));
  k_gemm_wmma<<<gg, gb, 0, stream>>>(Xq, Wtq, bq, Qh, 0);
  k_gemm_wmma<<<gg, gb, 0, stream>>>(Xk, Wtk, bk, Kh, 0);
  k_gemm_wmma<<<gg, gb, 0, stream>>>(Xv, Wtv, bv, Vh, 1);

  // 4) flash attention (WMMA)
  dim3 ab(128), ag(S_LEN / 64, NHEAD, BATCH);
  k_flash_attn<<<ag, ab, 0, stream>>>(Qh, Kh, Vh, mask, AO);

  // 5) output projection (WMMA, f32 out + bias)
  k_gemm_wmma<<<gg, gb, 0, stream>>>(AO, Wto, bo, d_out, 2);
}
